// UniRNN_19439021981760
// MI455X (gfx1250) — matrix operand order repaired and checked by reading, not compiled, not run on hardware
//
#include <hip/hip_runtime.h>
#include <hip/hip_bf16.h>
#include <stdint.h>

#define N_BATCH 32
#define T_SEQ   512
#define H_DIM   1024
#define V_DIM   8192

typedef __attribute__((ext_vector_type(16))) __bf16 v16bf;
typedef __attribute__((ext_vector_type(8)))  float  v8f;

union AFrag { v16bf v; uint4 q[2]; };

typedef __attribute__((address_space(3))) void* lds_as_ptr;

__device__ __forceinline__ uint16_t f32_to_bf16(float f) {
    uint32_t u = __float_as_uint(f);
    u += 0x7FFFu + ((u >> 16) & 1u);   // round-to-nearest-even
    return (uint16_t)(u >> 16);
}

// ---- CDNA5 async global->LDS copy (ASYNCcnt path, inline asm) ------------
// GLOBAL_LOAD_ASYNC_TO_LDS_B128 (VGLOBAL op 98): per lane,
//   16 bytes copied from the global address to the LDS address
__device__ __forceinline__ void async_copy16(const void* g, void* l) {
    uint32_t loff = (uint32_t)(uintptr_t)(lds_as_ptr)l;   // LDS byte address
    asm volatile("global_load_async_to_lds_b128 %0, %1, off"
                 :: "v"(loff), "v"(g) : "memory");
}

__device__ __forceinline__ void wait_async0() {
    asm volatile("s_wait_asynccnt 0x0" ::: "memory");
}

// ---- fragment loaders ----------------------------------------------------
// 16x32 bf16 A fragment from row-major memory (global or LDS).
__device__ __forceinline__ v16bf load_a_rowmajor(const uint16_t* __restrict__ A,
                                                 int lda, int lane) {
    int m  = lane & 15;
    int kb = (lane >> 4) << 4;
    const uint16_t* p = A + (size_t)m * lda + kb;
    AFrag f;
    f.q[0] = *reinterpret_cast<const uint4*>(p);
    f.q[1] = *reinterpret_cast<const uint4*>(p + 8);
    return f.v;
}

// Pre-packed 32x16 bf16 B fragment: lane l holds column (l&15),
// K = (l>>4)*16 .. +15, 16 contiguous bf16 per lane.
__device__ __forceinline__ v16bf load_b_packed(const uint16_t* __restrict__ Bp,
                                               int lane) {
    return *reinterpret_cast<const v16bf*>(Bp + lane * 16);
}

// ---- weight repack to B-fragment tile layout -----------------------------
__global__ void pack_b_bf16(const float* __restrict__ W, uint16_t* __restrict__ P,
                            int K, int Ncols) {
    int tid = blockIdx.x * blockDim.x + threadIdx.x;
    if (tid >= K * Ncols) return;
    int tile = tid >> 9;
    int rem  = tid & 511;
    int lane = rem >> 4;
    int j    = rem & 15;
    int ntn  = Ncols >> 4;
    int kt   = tile / ntn;
    int nt   = tile - kt * ntn;
    int krow = kt * 32 + ((lane >> 4) << 4) + j;
    int ncol = nt * 16 + (lane & 15);
    P[tid] = f32_to_bf16(W[(size_t)krow * Ncols + ncol]);
}

__global__ void zero_u16(uint16_t* __restrict__ p, int n) {
    int i = blockIdx.x * blockDim.x + threadIdx.x;
    if (i < n) p[i] = 0;
}

// ---- RNN recurrence step (latency-bound; 1 wave per 16x16 tile) ----------
__global__ void __launch_bounds__(32)
rnn_step(const int* __restrict__ x, const float* __restrict__ embed,
         const uint16_t* __restrict__ whh_p, const float* __restrict__ bh,
         const uint16_t* __restrict__ h_in_bf, uint16_t* __restrict__ h_out_bf,
         float* __restrict__ h_f32, uint16_t* __restrict__ z_bf, int t) {
    const int lane = threadIdx.x;
    const int n0   = blockIdx.x * 16;
    const int m0   = blockIdx.y * 16;
    const int ntn  = H_DIM / 16;

    v8f acc = {0.f, 0.f, 0.f, 0.f, 0.f, 0.f, 0.f, 0.f};
    for (int kt = 0; kt < H_DIM / 32; ++kt) {
        v16bf a = load_a_rowmajor(h_in_bf + (size_t)m0 * H_DIM + kt * 32, H_DIM, lane);
        v16bf b = load_b_packed(whh_p + ((size_t)kt * ntn + (n0 >> 4)) * 512, lane);
        acc = __builtin_amdgcn_wmma_f32_16x16x32_bf16(false, a, false, b,
                                                      (short)0, acc, false, false);
    }

    const int col  = n0 + (lane & 15);
    const int mhi  = (lane >> 4) * 8;
    const float bb = bh[col];
#pragma unroll
    for (int r = 0; r < 8; ++r) {
        int n  = m0 + mhi + r;
        int xi = x[n * T_SEQ + t];
        float v = acc[r] + embed[(size_t)xi * H_DIM + col] + bb;
        float h = tanhf(v);
        h_f32[(size_t)n * H_DIM + col] = h;
        uint16_t hb = f32_to_bf16(h);
        h_out_bf[(size_t)n * H_DIM + col] = hb;
        z_bf[((size_t)n * T_SEQ + t) * H_DIM + col] = hb;
    }
}

// ---- dominant GEMM: logits = z @ Wout + bout -----------------------------
// 128-thread block (4 waves, 2x2), 128x128 block tile, 64x64 per wave.
// Double-buffered async-LDS staging: per stage 8KB A + 8KB B.
__global__ void __launch_bounds__(128)
out_gemm(const uint16_t* __restrict__ z_bf, const uint16_t* __restrict__ wout_p,
         const float* __restrict__ bout, float* __restrict__ out) {
    __shared__ __attribute__((aligned(64))) uint16_t lA[2][128 * 32]; // 8KB x2
    __shared__ __attribute__((aligned(64))) uint16_t lB[2][8 * 512];  // 8KB x2

    const int tid  = threadIdx.x;
    const int lane = tid & 31;
    const int w    = tid >> 5;        // wave id 0..3
    const int wm   = (w >> 1) * 64;   // wave row offset in block tile
    const int wn   = (w & 1) * 64;    // wave col offset in block tile
    const int n0   = blockIdx.x * 128;
    const int m0   = blockIdx.y * 128;
    const int ntn  = V_DIM / 16;      // 512

    // stage kt into buffer buf (all 128 threads cooperate; 8x b128 each)
    auto stage = [&](int buf, int kt) {
#pragma unroll
        for (int it = 0; it < 4; ++it) {
            int q   = tid + 128 * it;          // 0..511 chunks of 16B
            int row = q >> 2, c = q & 3;
            async_copy16(z_bf + (size_t)(m0 + row) * H_DIM + kt * 32 + c * 8,
                         &lA[buf][row * 32 + c * 8]);
        }
        const uint16_t* bsrc = wout_p + ((size_t)kt * ntn + (n0 >> 4)) * 512;
#pragma unroll
        for (int it = 0; it < 4; ++it) {
            int q = tid + 128 * it;
            async_copy16(bsrc + q * 8, &lB[buf][q * 8]);
        }
    };

    v8f acc[4][4];
#pragma unroll
    for (int i = 0; i < 4; ++i)
#pragma unroll
        for (int j = 0; j < 4; ++j)
            acc[i][j] = (v8f){0.f, 0.f, 0.f, 0.f, 0.f, 0.f, 0.f, 0.f};

    stage(0, 0);
    int buf = 0;
    for (int kt = 0; kt < H_DIM / 32; ++kt) {
        wait_async0();        // this wave's stage-kt writes landed in LDS
        __syncthreads();      // everyone's stage-kt landed; prior reads done
        if (kt + 1 < H_DIM / 32) stage(buf ^ 1, kt + 1);

        v16bf a[4], b[4];
#pragma unroll
        for (int i = 0; i < 4; ++i)
            a[i] = load_a_rowmajor(&lA[buf][(wm + 16 * i) * 32], 32, lane);
#pragma unroll
        for (int j = 0; j < 4; ++j)
            b[j] = load_b_packed(&lB[buf][((wn >> 4) + j) * 512], lane);
#pragma unroll
        for (int i = 0; i < 4; ++i)
#pragma unroll
            for (int j = 0; j < 4; ++j)
                acc[i][j] = __builtin_amdgcn_wmma_f32_16x16x32_bf16(
                    false, a[i], false, b[j], (short)0, acc[i][j], false, false);
        buf ^= 1;
    }

    const int cl  = lane & 15;
    const int mhi = (lane >> 4) * 8;
#pragma unroll
    for (int i = 0; i < 4; ++i) {
#pragma unroll
        for (int j = 0; j < 4; ++j) {
            int col  = n0 + wn + 16 * j + cl;
            float bb = bout[col];
#pragma unroll
            for (int r = 0; r < 8; ++r) {
                int row = m0 + wm + 16 * i + mhi + r;
                out[(size_t)row * V_DIM + col] = acc[i][j][r] + bb;
            }
        }
    }
}

// ---- in-place row softmax (row cached in 32KB LDS) -----------------------
__global__ void __launch_bounds__(256)
softmax_rows(float* __restrict__ out) {
    __shared__ float buf[V_DIM];
    __shared__ float red[256];
    const int tid = threadIdx.x;
    float* p = out + (size_t)blockIdx.x * V_DIM;

    float mx = -INFINITY;
    for (int i = tid; i < V_DIM; i += 256) {
        float v = p[i]; buf[i] = v; mx = fmaxf(mx, v);
    }
    red[tid] = mx; __syncthreads();
    for (int s = 128; s > 0; s >>= 1) {
        if (tid < s) red[tid] = fmaxf(red[tid], red[tid + s]);
        __syncthreads();
    }
    mx = red[0]; __syncthreads();

    float sum = 0.f;
    for (int i = tid; i < V_DIM; i += 256) {
        float e = __expf(buf[i] - mx); buf[i] = e; sum += e;
    }
    red[tid] = sum; __syncthreads();
    for (int s = 128; s > 0; s >>= 1) {
        if (tid < s) red[tid] += red[tid + s];
        __syncthreads();
    }
    float inv = 1.0f / red[0];
    for (int i = tid; i < V_DIM; i += 256) p[i] = buf[i] * inv;
}

__global__ void copy_hlast(const float* __restrict__ h, float* __restrict__ dst, int n) {
    int i = blockIdx.x * blockDim.x + threadIdx.x;
    if (i < n) dst[i] = h[i];
}

extern "C" void kernel_launch(void* const* d_in, const int* in_sizes, int n_in,
                              void* d_out, int out_size, void* d_ws, size_t ws_size,
                              hipStream_t stream) {
    const int*   x     = (const int*)d_in[0];
    const float* embed = (const float*)d_in[1];
    const float* Whh   = (const float*)d_in[2];
    const float* bh    = (const float*)d_in[3];
    const float* Wout  = (const float*)d_in[4];
    const float* bout  = (const float*)d_in[5];
    float* y = (float*)d_out;

    char* ws = (char*)d_ws;
    uint16_t* whh_p  = (uint16_t*)(ws);                      //  2 MB
    uint16_t* wout_p = (uint16_t*)(ws + 2097152);            // 16 MB
    uint16_t* hbfA   = (uint16_t*)(ws + 18874368);           // 64 KB
    uint16_t* hbfB   = (uint16_t*)(ws + 18939904);           // 64 KB
    float*    h_f32  = (float*)   (ws + 19005440);           // 128 KB
    uint16_t* z_bf   = (uint16_t*)(ws + 19136512);           // 32 MB

    pack_b_bf16<<<(H_DIM * H_DIM + 255) / 256, 256, 0, stream>>>(Whh, whh_p, H_DIM, H_DIM);
    pack_b_bf16<<<(H_DIM * V_DIM + 255) / 256, 256, 0, stream>>>(Wout, wout_p, H_DIM, V_DIM);

    zero_u16<<<(N_BATCH * H_DIM + 255) / 256, 256, 0, stream>>>(hbfA, N_BATCH * H_DIM);

    for (int t = 0; t < T_SEQ; ++t) {
        const uint16_t* hin  = (t & 1) ? hbfB : hbfA;
        uint16_t*       hout = (t & 1) ? hbfA : hbfB;
        rnn_step<<<dim3(H_DIM / 16, N_BATCH / 16), 32, 0, stream>>>(
            x, embed, whh_p, bh, hin, hout, h_f32, z_bf, t);
    }

    out_gemm<<<dim3(V_DIM / 128, (N_BATCH * T_SEQ) / 128), 128, 0, stream>>>(
        z_bf, wout_p, bout, y);

    softmax_rows<<<N_BATCH * T_SEQ, 256, 0, stream>>>(y);

    copy_hlast<<<(N_BATCH * H_DIM + 255) / 256, 256, 0, stream>>>(
        h_f32, y + (size_t)N_BATCH * T_SEQ * V_DIM, N_BATCH * H_DIM);
}
